// DeformConv2d_90237262889598
// MI455X (gfx1250) — matrix operand order repaired and checked by reading, not compiled, not run on hardware
//
#include <hip/hip_runtime.h>

typedef __attribute__((ext_vector_type(16))) _Float16 v16h;
typedef __attribute__((ext_vector_type(8)))  _Float16 v8h;
typedef __attribute__((ext_vector_type(8)))  float    v8f;

#define B_    8
#define C_    64
#define H_    128
#define W_    128
#define HW_   (H_*W_)
#define G_    4
#define K_    9
#define COUT_ 64
#define CPG_  16
#define KK_   144      // CPG_*K_ (deform GEMM reduction dim)
#define KKP_  160      // padded to 5 chunks of 32
#define MOFF_ 72       // GD*K*2 offset channels
#define MOFFP_ 80      // padded to 16-multiple
#define MMSK_ 36       // GD*K mask channels
#define MMSKP_ 48      // padded
#define VSTRIDE_ 168   // LDS row stride in halves (336B, 16B aligned, conflict-breaking)

// ---------------- K1: fused depthwise 7x7 (both branches) + bias --------------
__global__ void dw7_both(const float* __restrict__ x,
                         const float* __restrict__ w0, const float* __restrict__ bb0,
                         const float* __restrict__ w1, const float* __restrict__ bb1,
                         float* __restrict__ t0, float* __restrict__ t1) {
  int idx = blockIdx.x * 256 + threadIdx.x;
  if (idx >= B_ * C_ * HW_) return;
  int w = idx % W_;
  int h = (idx / W_) % H_;
  int c = (idx / HW_) % C_;
  int b = idx / (C_ * HW_);
  const float* xp = x + ((size_t)(b * C_ + c)) * HW_;
  if (h + 4 < H_) __builtin_prefetch(xp + (h + 4) * W_ + w, 0, 0);  // global_prefetch_b8
  float s0 = bb0[c], s1 = bb1[c];
  #pragma unroll
  for (int dy = 0; dy < 7; ++dy) {
    int yy = h + dy - 3;
    if ((unsigned)yy >= (unsigned)H_) continue;
    #pragma unroll
    for (int dx = 0; dx < 7; ++dx) {
      int xx = w + dx - 3;
      if ((unsigned)xx >= (unsigned)W_) continue;
      float xv = xp[yy * W_ + xx];
      s0 = fmaf(xv, w0[c * 49 + dy * 7 + dx], s0);
      s1 = fmaf(xv, w1[c * 49 + dy * 7 + dx], s1);
    }
  }
  t0[idx] = s0;
  t1[idx] = s1;
}

// ---------------- K2: GroupNorm(1 group) stats per sample ---------------------
__global__ void gn_stats(const float* __restrict__ t, float* __restrict__ st) {
  __shared__ float s1[256], s2[256];
  int b = blockIdx.x;
  const float* p = t + (size_t)b * C_ * HW_;
  float a = 0.f, q = 0.f;
  for (int i = threadIdx.x; i < C_ * HW_; i += 256) {
    float v = p[i];
    a += v;
    q += v * v;
  }
  s1[threadIdx.x] = a;
  s2[threadIdx.x] = q;
  __syncthreads();
  for (int off = 128; off > 0; off >>= 1) {
    if (threadIdx.x < off) {
      s1[threadIdx.x] += s1[threadIdx.x + off];
      s2[threadIdx.x] += s2[threadIdx.x + off];
    }
    __syncthreads();
  }
  if (threadIdx.x == 0) {
    float n = (float)(C_ * HW_);
    float mean = s1[0] / n;
    float var = s2[0] / n - mean * mean;
    st[b * 2 + 0] = mean;
    st[b * 2 + 1] = rsqrtf(var + 1e-5f);
  }
}

// ------- K3: normalize + affine + transpose -> f16 [b][pixel][64] (WMMA B) ----
__global__ void __launch_bounds__(256)
gn_norm_t(const float* __restrict__ t, const float* __restrict__ st,
          const float* __restrict__ g, const float* __restrict__ bb,
          _Float16* __restrict__ xn_t) {
  __shared__ float tile[32][33];
  int b = blockIdx.z;
  int cb = blockIdx.y;   // channel tile (0..1)
  int hb = blockIdx.x;   // pixel tile (0..511)
  int tx = threadIdx.x & 31, ty = threadIdx.x >> 5;
  float mean = st[b * 2 + 0], rstd = st[b * 2 + 1];
  #pragma unroll
  for (int i = 0; i < 4; ++i) {
    int c = cb * 32 + ty + i * 8;
    tile[ty + i * 8][tx] =
        (t[((size_t)(b * C_ + c)) * HW_ + hb * 32 + tx] - mean) * rstd * g[c] + bb[c];
  }
  __syncthreads();
  #pragma unroll
  for (int i = 0; i < 4; ++i) {
    int row = ty + i * 8;  // pixel within tile
    xn_t[((size_t)b * HW_ + hb * 32 + row) * C_ + cb * 32 + tx] = (_Float16)tile[tx][row];
  }
}

// ------- pack: pointwise weights -> zero-padded f16 [Mpad][64] + padded bias --
__global__ void pack_pw(const float* __restrict__ w, const float* __restrict__ bias,
                        _Float16* __restrict__ w16, float* __restrict__ bpad,
                        int Mout, int Mpad) {
  int idx = blockIdx.x * 256 + threadIdx.x;
  if (idx < Mpad * C_) {
    int m = idx / C_, k = idx % C_;
    w16[idx] = (m < Mout) ? (_Float16)w[m * C_ + k] : (_Float16)0.f;
  }
  if (idx < Mpad) bpad[idx] = (idx < Mout) ? bias[idx] : 0.f;
}

// ------- pack: deform conv weights -> zero-padded f16 [64][160], ck = c*9+k ---
__global__ void pack_dw(const float* __restrict__ w, _Float16* __restrict__ w16) {
  int idx = blockIdx.x * 256 + threadIdx.x;
  if (idx >= COUT_ * KKP_) return;
  int m = idx / KKP_, ck = idx % KKP_;
  float v = 0.f;
  if (ck < KK_) v = w[(m * CPG_ + ck / K_) * K_ + ck % K_];
  w16[idx] = (_Float16)v;
}

// ---------------- K4: 1x1 conv as WMMA GEMM: D[M,Npix] = W[M,64] x Xn[64,Npix]
// Branch-free: A from packed f16 weights (2x b128), B from pixel-major f16 (2x b128).
__global__ void __launch_bounds__(256)
pw_wmma(const _Float16* __restrict__ xn_t, const _Float16* __restrict__ w16,
        const float* __restrict__ bpad, float* __restrict__ out, int Mpad) {
  int b = blockIdx.z;
  int m0 = blockIdx.y * 16;
  int wv = threadIdx.x >> 5;
  int lane = threadIdx.x & 31;
  int r16 = lane & 15;       // A row (M), B col (N), D col (N)
  int hi = lane >> 4;
  int p = blockIdx.x * 128 + wv * 16 + r16;
  const _Float16* xrow = xn_t + ((size_t)b * HW_ + p) * C_;
  const _Float16* wrow = w16 + (size_t)(m0 + r16) * C_;
  v8f acc = {};
  #pragma unroll
  for (int kc = 0; kc < 2; ++kc) {
    int k0 = kc * 32;
    // A 16x32 layout: lane holds K = k0+8*hi+{0..7} then +16 (two contiguous runs)
    v8h a0 = *(const v8h*)(wrow + k0 + hi * 8);
    v8h a1 = *(const v8h*)(wrow + k0 + hi * 8 + 16);
    v16h av = __builtin_shufflevector(a0, a1, 0, 1, 2, 3, 4, 5, 6, 7,
                                      8, 9, 10, 11, 12, 13, 14, 15);
    // B 32x16 layout: lane holds K = k0+8*hi+{0..7} then +16 (matches A)
    v8h b0 = *(const v8h*)(xrow + k0 + hi * 8);
    v8h b1 = *(const v8h*)(xrow + k0 + hi * 8 + 16);
    v16h bv = __builtin_shufflevector(b0, b1, 0, 1, 2, 3, 4, 5, 6, 7,
                                      8, 9, 10, 11, 12, 13, 14, 15);
    acc = __builtin_amdgcn_wmma_f32_16x16x32_f16(false, av, false, bv,
                                                 (short)0, acc, false, false);
  }
  #pragma unroll
  for (int r = 0; r < 8; ++r) {
    int mo = m0 + r + hi * 8;
    out[((size_t)b * Mpad + mo) * HW_ + p] = acc[r] + bpad[mo];
  }
}

// ---------------- K5: softmax over the 9 kernel taps (in place, padded stride)
__global__ void softmax_k(float* __restrict__ m) {
  int idx = blockIdx.x * 256 + threadIdx.x;
  if (idx >= B_ * G_ * HW_) return;
  int hw = idx % HW_;
  int gd = (idx / HW_) % G_;
  int b = idx / (G_ * HW_);
  float* p = m + ((size_t)b * MMSKP_ + gd * K_) * HW_ + hw;
  float v[K_];
  float mx = -1e30f;
  #pragma unroll
  for (int k = 0; k < K_; ++k) { v[k] = p[(size_t)k * HW_]; mx = fmaxf(mx, v[k]); }
  float s = 0.f;
  #pragma unroll
  for (int k = 0; k < K_; ++k) { v[k] = __expf(v[k] - mx); s += v[k]; }
  float inv = 1.f / s;
  #pragma unroll
  for (int k = 0; k < K_; ++k) p[(size_t)k * HW_] = v[k] * inv;
}

// ---------------- K6: deformable sampling + grouped 3x3 conv via WMMA ---------
// Block = one (b, group, image row). Sample tile stored pixel-major in LDS so the
// phase-2 B operand is two ds_load_b128 per chunk.
__global__ void __launch_bounds__(256)
deform_wmma(const float* __restrict__ x, const float* __restrict__ offb,
            const float* __restrict__ mb, const _Float16* __restrict__ wd16,
            const float* __restrict__ bias, float* __restrict__ out) {
  __shared__ _Float16 Vt[128 * VSTRIDE_];   // [pixel][ck] 42 KB
  __shared__ int      cidx[K_ * 128 * 2];   // packed clamped (y<<16|x) pairs
  __shared__ _Float16 cw[K_ * 128 * 4];     // 4 bilinear weights * mask
  int ho = blockIdx.x;
  int g  = blockIdx.y;
  int b  = blockIdx.z;
  int tid = threadIdx.x;

  // phase 0: per-(tap, pixel) sampling geometry
  for (int i = tid; i < K_ * 128; i += 256) {
    int k = i >> 7;
    int n = i & 127;
    int hw = ho * W_ + n;
    int ch = (g * K_ + k) * 2;
    float offy = offb[((size_t)b * MOFFP_ + ch + 0) * HW_ + hw];
    float offx = offb[((size_t)b * MOFFP_ + ch + 1) * HW_ + hw];
    float mval = mb[((size_t)b * MMSKP_ + g * K_ + k) * HW_ + hw];
    float py = (float)(ho - 1 + k / 3) + offy;
    float px = (float)(n - 1 + k % 3) + offx;
    float fy = floorf(py), fx = floorf(px);
    int y0 = (int)fy, x0 = (int)fx;
    int y1 = y0 + 1, x1 = x0 + 1;
    float wy = py - fy, wx = px - fx;
    float vy0 = (y0 >= 0 && y0 < H_) ? 1.f : 0.f;
    float vy1 = (y1 >= 0 && y1 < H_) ? 1.f : 0.f;
    float vx0 = (x0 >= 0 && x0 < W_) ? 1.f : 0.f;
    float vx1 = (x1 >= 0 && x1 < W_) ? 1.f : 0.f;
    int yc0 = min(max(y0, 0), H_ - 1), yc1 = min(max(y1, 0), H_ - 1);
    int xc0 = min(max(x0, 0), W_ - 1), xc1 = min(max(x1, 0), W_ - 1);
    cidx[i * 2 + 0] = (yc0 << 16) | xc0;
    cidx[i * 2 + 1] = (yc1 << 16) | xc1;
    cw[i * 4 + 0] = (_Float16)((1.f - wy) * (1.f - wx) * vy0 * vx0 * mval);
    cw[i * 4 + 1] = (_Float16)((1.f - wy) * wx * vy0 * vx1 * mval);
    cw[i * 4 + 2] = (_Float16)(wy * (1.f - wx) * vy1 * vx0 * mval);
    cw[i * 4 + 3] = (_Float16)(wy * wx * vy1 * vx1 * mval);
  }
  // zero-fill the K padding region (ck = 144..159) read by the last WMMA chunk
  for (int i = tid; i < 128 * (KKP_ - KK_); i += 256) {
    int n = i >> 4;
    Vt[n * VSTRIDE_ + KK_ + (i & 15)] = (_Float16)0.f;
  }
  __syncthreads();

  // phase 1: modulated bilinear gather -> f16 LDS tile Vt[n][ck], ck = c*9+k
  for (int i = tid; i < KK_ * 128; i += 256) {
    int ck = i >> 7;
    int n = i & 127;
    int c = ck / K_;
    int k = ck - c * K_;
    int ci = k * 128 + n;
    int pa = cidx[ci * 2 + 0], pb = cidx[ci * 2 + 1];
    int y0 = pa >> 16, x0c = pa & 0xffff;
    int y1 = pb >> 16, x1c = pb & 0xffff;
    const float* xp = x + ((size_t)(b * C_) + g * CPG_ + c) * HW_;
    float w00 = (float)cw[ci * 4 + 0], w01 = (float)cw[ci * 4 + 1];
    float w10 = (float)cw[ci * 4 + 2], w11 = (float)cw[ci * 4 + 3];
    float v = w00 * xp[y0 * W_ + x0c] + w01 * xp[y0 * W_ + x1c]
            + w10 * xp[y1 * W_ + x0c] + w11 * xp[y1 * W_ + x1c];
    Vt[n * VSTRIDE_ + ck] = (_Float16)v;
  }
  __syncthreads();

  // phase 2: WMMA GEMM, K = 144 padded to 160 (5 chunks of 32), branch-free
  int wv = tid >> 5;
  int lane = tid & 31;
  int r16 = lane & 15;
  int hi = lane >> 4;
  const _Float16* wrow = wd16 + (size_t)(g * CPG_ + r16) * KKP_;
  const _Float16* vrow = &Vt[(wv * 16 + r16) * VSTRIDE_];
  v8f acc = {};
  #pragma unroll
  for (int kc = 0; kc < 5; ++kc) {
    int k0 = kc * 32;
    v8h a0 = *(const v8h*)(wrow + k0 + hi * 8);
    v8h a1 = *(const v8h*)(wrow + k0 + hi * 8 + 16);
    v16h av = __builtin_shufflevector(a0, a1, 0, 1, 2, 3, 4, 5, 6, 7,
                                      8, 9, 10, 11, 12, 13, 14, 15);
    v8h b0 = *(const v8h*)(vrow + k0 + hi * 8);
    v8h b1 = *(const v8h*)(vrow + k0 + hi * 8 + 16);
    v16h bv = __builtin_shufflevector(b0, b1, 0, 1, 2, 3, 4, 5, 6, 7,
                                      8, 9, 10, 11, 12, 13, 14, 15);
    acc = __builtin_amdgcn_wmma_f32_16x16x32_f16(false, av, false, bv,
                                                 (short)0, acc, false, false);
  }
  #pragma unroll
  for (int r = 0; r < 8; ++r) {
    int mo = g * CPG_ + r + hi * 8;
    out[((size_t)b * COUT_ + mo) * HW_ + ho * W_ + wv * 16 + r16] = acc[r] + bias[mo];
  }
}

// -----------------------------------------------------------------------------
extern "C" void kernel_launch(void* const* d_in, const int* in_sizes, int n_in,
                              void* d_out, int out_size, void* d_ws, size_t ws_size,
                              hipStream_t stream) {
  (void)in_sizes; (void)n_in; (void)out_size; (void)ws_size;
  const float* x        = (const float*)d_in[0];
  const float* w_dw_off = (const float*)d_in[1];
  const float* b_dw_off = (const float*)d_in[2];
  const float* gn_off_g = (const float*)d_in[3];
  const float* gn_off_b = (const float*)d_in[4];
  const float* w_pw_off = (const float*)d_in[5];
  const float* b_pw_off = (const float*)d_in[6];
  const float* w_dw_m   = (const float*)d_in[7];
  const float* b_dw_m   = (const float*)d_in[8];
  const float* gn_m_g   = (const float*)d_in[9];
  const float* gn_m_b   = (const float*)d_in[10];
  const float* w_pw_m   = (const float*)d_in[11];
  const float* b_pw_m   = (const float*)d_in[12];
  const float* weight   = (const float*)d_in[13];
  const float* bias     = (const float*)d_in[14];
  float* out = (float*)d_out;

  char* ws = (char*)d_ws;
  size_t o = 0;
  auto align256 = [&](size_t v) { return (v + 255) & ~(size_t)255; };
  float*    t_off  = (float*)(ws + o);    o = align256(o + (size_t)B_ * C_ * HW_ * 4);
  float*    t_m    = (float*)(ws + o);    o = align256(o + (size_t)B_ * C_ * HW_ * 4);
  float*    stats  = (float*)(ws + o);    o = align256(o + 256);
  _Float16* xn_off = (_Float16*)(ws + o); o = align256(o + (size_t)B_ * C_ * HW_ * 2);
  _Float16* xn_m   = (_Float16*)(ws + o); o = align256(o + (size_t)B_ * C_ * HW_ * 2);
  float*    offb   = (float*)(ws + o);    o = align256(o + (size_t)B_ * MOFFP_ * HW_ * 4);
  float*    mbuf   = (float*)(ws + o);    o = align256(o + (size_t)B_ * MMSKP_ * HW_ * 4);
  _Float16* w16o   = (_Float16*)(ws + o); o = align256(o + (size_t)MOFFP_ * C_ * 2);
  _Float16* w16m   = (_Float16*)(ws + o); o = align256(o + (size_t)MMSKP_ * C_ * 2);
  float*    bpo    = (float*)(ws + o);    o = align256(o + (size_t)MOFFP_ * 4);
  float*    bpm    = (float*)(ws + o);    o = align256(o + (size_t)MMSKP_ * 4);
  _Float16* wd16   = (_Float16*)(ws + o); o = align256(o + (size_t)COUT_ * KKP_ * 2);

  int total = B_ * C_ * HW_;
  dw7_both<<<(total + 255) / 256, 256, 0, stream>>>(x, w_dw_off, b_dw_off,
                                                    w_dw_m, b_dw_m, t_off, t_m);
  gn_stats<<<B_, 256, 0, stream>>>(t_off, stats);
  gn_stats<<<B_, 256, 0, stream>>>(t_m, stats + 16);
  gn_norm_t<<<dim3(HW_ / 32, C_ / 32, B_), 256, 0, stream>>>(t_off, stats,
                                                             gn_off_g, gn_off_b, xn_off);
  gn_norm_t<<<dim3(HW_ / 32, C_ / 32, B_), 256, 0, stream>>>(t_m, stats + 16,
                                                             gn_m_g, gn_m_b, xn_m);

  pack_pw<<<(MOFFP_ * C_ + 255) / 256, 256, 0, stream>>>(w_pw_off, b_pw_off, w16o, bpo,
                                                         MOFF_, MOFFP_);
  pack_pw<<<(MMSKP_ * C_ + 255) / 256, 256, 0, stream>>>(w_pw_m, b_pw_m, w16m, bpm,
                                                         MMSK_, MMSKP_);
  pack_dw<<<(COUT_ * KKP_ + 255) / 256, 256, 0, stream>>>(weight, wd16);

  // pointwise convs as WMMA GEMMs: grid = (pixel tiles of 128, M tiles of 16, B)
  pw_wmma<<<dim3(HW_ / 128, MOFFP_ / 16, B_), 256, 0, stream>>>(xn_off, w16o, bpo, offb, MOFFP_);
  pw_wmma<<<dim3(HW_ / 128, MMSKP_ / 16, B_), 256, 0, stream>>>(xn_m, w16m, bpm, mbuf, MMSKP_);

  int tsm = B_ * G_ * HW_;
  softmax_k<<<(tsm + 255) / 256, 256, 0, stream>>>(mbuf);

  deform_wmma<<<dim3(H_, G_, B_), 256, 0, stream>>>(x, offb, mbuf, wd16, bias, out);
}
